// DualStateAttention_21672404976162
// MI455X (gfx1250) — compile-verified, matrix operand order checked by reading, not run on hardware
//
#include <hip/hip_runtime.h>
#include <hip/hip_bf16.h>

// ---------------------------------------------------------------------------
// DualStateAttention on gfx1250 (MI455X):
//   C0: one-time f32->bf16 convert of x; transpose+convert of w_qkv, w_out
//   K1: qkv = x @ w_qkv   (bf16 WMMA, f32 accum, double-buffered async-LDS)
//   S1: segmented scan pass 1 (per-segment zero-init summaries, 16 segs)
//   S2: chain summaries + replay segment, write bf16 out_cat + final states
//   K3: out = concat(out_f,out_s) @ w_out
// ---------------------------------------------------------------------------

typedef __attribute__((ext_vector_type(16))) __bf16 v16bf;
typedef __attribute__((ext_vector_type(8)))  float  v8f;

#define BATCH 4
#define SEQ   4096
#define HID   1024
#define NHEAD 16
#define HDIM  64
#define MROWS (BATCH * SEQ)          // 16384
#define SEGS  16
#define CSEG  (SEQ / SEGS)           // 256

union Frag16 {
    v16bf v;
    uint4 q[2];
};

static __device__ __forceinline__ unsigned short f2bf(float f) {
    unsigned int u = __float_as_uint(f);
    unsigned int r = u + 0x7FFFu + ((u >> 16) & 1u);
    return (unsigned short)(r >> 16);
}

// async global -> LDS copy of 16 bytes (CDNA5 async path, ASYNCcnt)
static __device__ __forceinline__ void async_ld_b128(unsigned lds_off,
                                                     const unsigned short* g) {
    asm volatile("global_load_async_to_lds_b128 %0, %1, off"
                 :: "v"(lds_off), "v"(g) : "memory");
}
static __device__ __forceinline__ void wait_async0() {
    asm volatile("s_wait_asynccnt 0x0" ::: "memory");
}

// ---------------------------------------------------------------------------
// One-time converts
// ---------------------------------------------------------------------------
__global__ void dsa_cvt_bf16(const float* __restrict__ in,
                             unsigned short* __restrict__ out, int n) {
    int i = blockIdx.x * blockDim.x + threadIdx.x;
    if (i < n) out[i] = f2bf(in[i]);
}

// in [K,N] f32 -> out [N,K] bf16 (transposed, K-major for WMMA B)
__global__ void dsa_cvt_t_bf16(const float* __restrict__ in,
                               unsigned short* __restrict__ out, int K, int N) {
    int i = blockIdx.x * blockDim.x + threadIdx.x;
    if (i < N * K) {
        int n = i / K, k = i - n * K;
        out[i] = f2bf(in[(size_t)k * N + n]);
    }
}

// ---------------------------------------------------------------------------
// C[M,N] = A[M,K] * Bt[N,K]^T ; A,Bt bf16 row-major (K fastest), C f32.
// Block tile 128x128, K-stage 64, double-buffered LDS with async copies
// overlapping the 16 WMMAs of the current stage.
// 8 waves: wm(2) x wn(4); wave tile 64x32 = 8 accumulators.
// LDS rows padded to 72 ushorts (144B) to spread ds_load_b128 banks.
// ---------------------------------------------------------------------------
#define LDS_STRIDE 72
#define TILE_BYTES (128 * LDS_STRIDE * 2)   // 18432 B per tile buffer

__global__ __launch_bounds__(256) void dsa_gemm_bf16_wmma(
    const unsigned short* __restrict__ A, const unsigned short* __restrict__ Bt,
    float* __restrict__ C, int M, int N, int K)
{
    __shared__ __align__(16) unsigned short sA[2][128 * LDS_STRIDE];
    __shared__ __align__(16) unsigned short sB[2][128 * LDS_STRIDE];

    const int tid   = threadIdx.x;
    const int m_blk = blockIdx.y * 128;
    const int n_blk = blockIdx.x * 128;
    const int wid   = tid >> 5;
    const int lane  = tid & 31;
    const int wm    = wid >> 2;           // 0..1 : 64 rows each
    const int wn    = wid & 3;            // 0..3 : 32 cols each

    const unsigned baseA = (unsigned)(unsigned long long)&sA[0][0];
    const unsigned baseB = (unsigned)(unsigned long long)&sB[0][0];
    const char* sAb = (const char*)&sA[0][0];
    const char* sBb = (const char*)&sB[0][0];

    v8f acc[8];
#pragma unroll
    for (int i = 0; i < 8; ++i)
        acc[i] = (v8f){0.f, 0.f, 0.f, 0.f, 0.f, 0.f, 0.f, 0.f};

    const int l    = lane & 15;
    const int koff = (lane >> 4) ? 8 : 0;   // ISA 16-bit A/B lane K split

    // stage a 128x64 bf16 tile pair into buffer `buf` (4 b128 chunks/thread ea.)
    auto stage = [&](int k0, int buf) {
        const unsigned bo = (unsigned)buf * TILE_BYTES;
#pragma unroll
        for (int c = tid; c < 1024; c += 256) {
            const int row  = c >> 3;
            const int part = c & 7;
            async_ld_b128(baseA + bo + row * 144 + part * 16,
                          A + (size_t)(m_blk + row) * K + k0 + part * 8);
            async_ld_b128(baseB + bo + row * 144 + part * 16,
                          Bt + (size_t)(n_blk + row) * K + k0 + part * 8);
        }
    };

    // prologue: fill buffer 0
    stage(0, 0);
    wait_async0();
    __syncthreads();

    int cur = 0;
    for (int k0 = 0; k0 < K; k0 += 64) {
        // kick off next stage into the other buffer (overlaps WMMAs below)
        if (k0 + 64 < K) stage(k0 + 64, cur ^ 1);

        const int cb = cur * TILE_BYTES;
#pragma unroll
        for (int ks = 0; ks < 2; ++ks) {
            const int kb = cb + (ks * 32 + koff) * 2;   // byte offset in row
            Frag16 af[4], bf[2];
#pragma unroll
            for (int i = 0; i < 4; ++i) {
                const int byte = (wm * 64 + i * 16 + l) * 144 + kb;
                af[i].q[0] = *(const uint4*)(sAb + byte);
                af[i].q[1] = *(const uint4*)(sAb + byte + 32);
            }
#pragma unroll
            for (int j = 0; j < 2; ++j) {
                const int byte = (wn * 32 + j * 16 + l) * 144 + kb;
                bf[j].q[0] = *(const uint4*)(sBb + byte);
                bf[j].q[1] = *(const uint4*)(sBb + byte + 32);
            }
#pragma unroll
            for (int i = 0; i < 4; ++i)
#pragma unroll
                for (int j = 0; j < 2; ++j)
                    acc[i * 2 + j] = __builtin_amdgcn_wmma_f32_16x16x32_bf16(
                        false, af[i].v, false, bf[j].v, (short)0,
                        acc[i * 2 + j], false, false);
        }

        wait_async0();      // my wave's next-stage copies have landed
        __syncthreads();    // everyone done reading `cur` + copies visible
        cur ^= 1;
    }

    // ---- epilogue: 16x16 f32 C layout (lanes 0-15: M=v; 16-31: M=v+8) ----
    const int mo = (lane >> 4) * 8;
#pragma unroll
    for (int i = 0; i < 4; ++i) {
        const int mbase = m_blk + wm * 64 + i * 16 + mo;
#pragma unroll
        for (int j = 0; j < 2; ++j) {
            const int cn = n_blk + wn * 32 + j * 16 + l;
#pragma unroll
            for (int v = 0; v < 8; ++v)
                C[(size_t)(mbase + v) * N + cn] = acc[i * 2 + j][v];
        }
    }
}

// ---------------------------------------------------------------------------
// Segmented linear scan. One wave32 per (seg, mode, b, h); lane owns d, d+32.
// ---------------------------------------------------------------------------
static __device__ __forceinline__ float phi_elu1(float x) {
    return x > 0.f ? x + 1.f : __expf(x);
}

static __device__ __forceinline__ void decode_scan_idx(
    int idx, int& seg, int& mode, int& b, int& h) {
    seg  = idx % SEGS;
    int r = idx / SEGS;
    mode = r & 1;
    int bh = r >> 1;
    b = bh / NHEAD;
    h = bh % NHEAD;
}

// Pass 1: zero-init segment recurrence; store only final (kv, ks) summaries.
__global__ __launch_bounds__(32) void dsa_scan_p1(
    const float* __restrict__ qkv, const float* __restrict__ alpha,
    const float* __restrict__ beta, float* __restrict__ sums)
{
    int seg, mode, b, h;
    decode_scan_idx(blockIdx.x, seg, mode, b, h);
    const int lane = threadIdx.x, d0 = lane, d1 = lane + 32;

    const float gate = mode ? beta[h] : alpha[h];
    const float dcay = fmaxf(1e-8f, 1.f / (1.f + __expf(-gate)));

    float kv0 = 0.f, kv1 = 0.f, ks0 = 0.f, ks1 = 0.f;
    const int t0 = seg * CSEG;
    for (int t = t0; t < t0 + CSEG; ++t) {
        const size_t base = ((size_t)(b * SEQ + t)) * (3 * HID) + h * HDIM;
        const float k0 = phi_elu1(qkv[base + HID + d0]);
        const float k1 = phi_elu1(qkv[base + HID + d1]);
        const float v0 = qkv[base + 2 * HID + d0];
        const float v1 = qkv[base + 2 * HID + d1];
        kv0 = fmaf(dcay, kv0, k0 * v0);
        kv1 = fmaf(dcay, kv1, k1 * v1);
        ks0 = fmaf(dcay, ks0, k0);
        ks1 = fmaf(dcay, ks1, k1);
    }
    const size_t s = ((((size_t)mode * BATCH + b) * NHEAD + h) * SEGS + seg) * 128;
    sums[s + d0] = kv0;  sums[s + d1] = kv1;
    sums[s + 64 + d0] = ks0;  sums[s + 64 + d1] = ks1;
}

// Pass 2: chain summaries -> segment init; replay segment producing outputs.
__global__ __launch_bounds__(32) void dsa_scan_p2(
    const float* __restrict__ qkv, const float* __restrict__ alpha,
    const float* __restrict__ beta, const float* __restrict__ sums,
    unsigned short* __restrict__ out_cat, float* __restrict__ dout)
{
    int seg, mode, b, h;
    decode_scan_idx(blockIdx.x, seg, mode, b, h);
    const int lane = threadIdx.x, d0 = lane, d1 = lane + 32;

    const float gate = mode ? beta[h] : alpha[h];
    const float dcay = fmaxf(1e-8f, 1.f / (1.f + __expf(-gate)));
    const float dC   = __powf(dcay, (float)CSEG);

    // init = chained end-state of previous segments (linear recurrence)
    float kv0 = 0.f, kv1 = 0.f, ks0 = 0.f, ks1 = 0.f;
    const size_t sb = (((size_t)mode * BATCH + b) * NHEAD + h) * SEGS;
    for (int j = 0; j < seg; ++j) {
        const float* sj = sums + (sb + j) * 128;
        kv0 = fmaf(dC, kv0, sj[d0]);
        kv1 = fmaf(dC, kv1, sj[d1]);
        ks0 = fmaf(dC, ks0, sj[64 + d0]);
        ks1 = fmaf(dC, ks1, sj[64 + d1]);
    }

    const int t0 = seg * CSEG;
    for (int t = t0; t < t0 + CSEG; ++t) {
        const size_t base = ((size_t)(b * SEQ + t)) * (3 * HID) + h * HDIM;
        const float q0 = phi_elu1(qkv[base + d0]);
        const float q1 = phi_elu1(qkv[base + d1]);
        const float k0 = phi_elu1(qkv[base + HID + d0]);
        const float k1 = phi_elu1(qkv[base + HID + d1]);
        const float v0 = qkv[base + 2 * HID + d0];
        const float v1 = qkv[base + 2 * HID + d1];

        kv0 = fmaf(dcay, kv0, k0 * v0);
        kv1 = fmaf(dcay, kv1, k1 * v1);
        ks0 = fmaf(dcay, ks0, k0);
        ks1 = fmaf(dcay, ks1, k1);

        float part = fmaf(q0, ks0, q1 * ks1);
#pragma unroll
        for (int off = 16; off > 0; off >>= 1)
            part += __shfl_xor(part, off, 32);
        const float inv = 1.f / fmaxf(part, 1e-6f);

        const size_t ob =
            (size_t)(b * SEQ + t) * (2 * HID) + (size_t)mode * HID + h * HDIM;
        out_cat[ob + d0] = f2bf(q0 * kv0 * inv);
        out_cat[ob + d1] = f2bf(q1 * kv1 * inv);
    }

    if (seg == SEGS - 1) {  // final states: (kv_f1, ks_f1, kv_s1, ks_s1)
        const size_t OUT_ELEMS = (size_t)MROWS * HID;
        const size_t SB = (size_t)BATCH * NHEAD * HDIM;
        float* st = dout + OUT_ELEMS + (size_t)mode * 2 * SB
                    + (size_t)(b * NHEAD + h) * HDIM;
        st[d0] = kv0;  st[d1] = kv1;
        st[SB + d0] = ks0;  st[SB + d1] = ks1;
    }
}

// ---------------------------------------------------------------------------
extern "C" void kernel_launch(void* const* d_in, const int* in_sizes, int n_in,
                              void* d_out, int out_size, void* d_ws, size_t ws_size,
                              hipStream_t stream) {
    (void)in_sizes; (void)n_in; (void)out_size; (void)ws_size;

    const float* x     = (const float*)d_in[0];   // [B,T,HID]
    const float* w_qkv = (const float*)d_in[1];   // [HID, 3*HID]
    const float* w_out = (const float*)d_in[2];   // [2*HID, HID]
    const float* alpha = (const float*)d_in[3];
    const float* beta  = (const float*)d_in[4];
    float*       out   = (float*)d_out;

    // workspace layout
    char* ws = (char*)d_ws;
    float*          qkv     = (float*)ws;                          // 192 MB f32
    unsigned short* xb      = (unsigned short*)(ws + 201326592);   //  32 MB bf16
    unsigned short* wqT     = (unsigned short*)(ws + 234881024);   //   6 MB bf16
    unsigned short* woT     = (unsigned short*)(ws + 241172480);   //   4 MB bf16
    unsigned short* out_cat = (unsigned short*)(ws + 245366784);   //  64 MB bf16
    float*          sums    = (float*)(ws + 312475648);            //   1 MB f32

    // C0: one-time converts / weight transposes
    dsa_cvt_bf16<<<(MROWS * HID) / 256, 256, 0, stream>>>(x, xb, MROWS * HID);
    dsa_cvt_t_bf16<<<(3 * HID * HID) / 256, 256, 0, stream>>>(w_qkv, wqT,
                                                              HID, 3 * HID);
    dsa_cvt_t_bf16<<<(2 * HID * HID) / 256, 256, 0, stream>>>(w_out, woT,
                                                              2 * HID, HID);

    // K1: qkv = x @ w_qkv   (M=16384, N=3072, K=1024)
    {
        dim3 grid((3 * HID) / 128, MROWS / 128);
        dsa_gemm_bf16_wmma<<<grid, 256, 0, stream>>>(xb, wqT, qkv,
                                                     MROWS, 3 * HID, HID);
    }
    // S1/S2: segmented dual-decay scan (2048 independent wave32 blocks each)
    const int nscan = SEGS * 2 * BATCH * NHEAD;
    dsa_scan_p1<<<nscan, 32, 0, stream>>>(qkv, alpha, beta, sums);
    dsa_scan_p2<<<nscan, 32, 0, stream>>>(qkv, alpha, beta, sums, out_cat, out);

    // K3: out = out_cat @ w_out   (M=16384, N=1024, K=2048)
    {
        dim3 grid(HID / 128, MROWS / 128);
        dsa_gemm_bf16_wmma<<<grid, 256, 0, stream>>>(out_cat, woT, out,
                                                     MROWS, HID, 2 * HID);
    }
}
